// GCN_34282428957176
// MI455X (gfx1250) — matrix operand order repaired and checked by reading, not compiled, not run on hardware
//
#include <hip/hip_runtime.h>
#include <hip/hip_bf16.h>
#include <stdint.h>

// ---------------------------------------------------------------------------
// CDNA5 (gfx1250) WMMA types, per CDNA5_HIP.md (probe-verified signatures)
// ---------------------------------------------------------------------------
typedef __attribute__((ext_vector_type(16))) __bf16 v16bf;
typedef __attribute__((ext_vector_type(8)))  float  v8f;

union BFrag {
  v16bf v;
  unsigned short u[16];
  uint4 q[2];
};

// fp32 -> bf16, round-to-nearest-even (only used in memory-bound prep passes)
__device__ __forceinline__ unsigned short f2bf(float x) {
  unsigned int b = __float_as_uint(x);
  unsigned int r = (b + 0x7FFFu + ((b >> 16) & 1u)) >> 16;
  return (unsigned short)r;
}

// ---------------------------------------------------------------------------
// Prep kernels (degree / rsqrt / conversions / fragment swizzles)
// ---------------------------------------------------------------------------
__global__ void fill1_kernel(float* __restrict__ p, int n) {
  int i = blockIdx.x * blockDim.x + threadIdx.x;
  if (i < n) p[i] = 1.0f;  // self-loop contributes 1 to in-degree
}

__global__ void deg_count_kernel(const int* __restrict__ col, float* __restrict__ deg, int E) {
  int e = blockIdx.x * blockDim.x + threadIdx.x;
  if (e < E) atomicAdd(&deg[col[e]], 1.0f);
}

__global__ void rsqrt_kernel(float* __restrict__ p, int n) {
  int i = blockIdx.x * blockDim.x + threadIdx.x;
  if (i < n) p[i] = rsqrtf(p[i]);
}

__global__ void cvt_bf16x4_kernel(const float* __restrict__ in, unsigned short* __restrict__ out,
                                  int n4) {
  int i = blockIdx.x * blockDim.x + threadIdx.x;
  if (i < n4) {
    float4 v = ((const float4*)in)[i];
    ushort4 o;
    o.x = f2bf(v.x); o.y = f2bf(v.y); o.z = f2bf(v.z); o.w = f2bf(v.w);
    ((ushort4*)out)[i] = o;
  }
}

__global__ void cvt_relu_bf16x4_kernel(const float* __restrict__ in, unsigned short* __restrict__ out,
                                       int n4) {
  int i = blockIdx.x * blockDim.x + threadIdx.x;
  if (i < n4) {
    float4 v = ((const float4*)in)[i];
    ushort4 o;
    o.x = f2bf(fmaxf(v.x, 0.f)); o.y = f2bf(fmaxf(v.y, 0.f));
    o.z = f2bf(fmaxf(v.z, 0.f)); o.w = f2bf(fmaxf(v.w, 0.f));
    ((ushort4*)out)[i] = o;
  }
}

// Build B panels in WMMA-fragment order (bf16):
//   dst[(((nt*KB + kb)*32 + lane)*16) + e] = bf16(W[k*C + n]),
//   n = nt*16 + (lane&15), k = kb*32 + (lane>>4)*16 + e, zero-pad when n >= C.
__global__ void build_bfrag_kernel(const float* __restrict__ W, unsigned short* __restrict__ dst,
                                   int C, int NT, int KB) {
  int idx = blockIdx.x * blockDim.x + threadIdx.x;
  if (idx >= NT * KB * 32 * 16) return;
  int e    = idx & 15;
  int lane = (idx >> 4) & 31;
  int f    = idx >> 9;             // nt*KB + kb
  int kb   = f % KB;
  int nt   = f / KB;
  int n = nt * 16 + (lane & 15);
  int k = kb * 32 + (lane >> 4) * 16 + e;
  float v = (n < C) ? W[k * C + n] : 0.f;
  dst[idx] = f2bf(v);
}

// ---------------------------------------------------------------------------
// CSR build (destination-major): count -> exclusive scan -> cursor scatter.
// Replaces ~134M scatter atomics with ~1.6M build atomics + gather aggregation.
// ---------------------------------------------------------------------------
__global__ void zero_u32_kernel(unsigned int* __restrict__ p, int n) {
  int i = blockIdx.x * blockDim.x + threadIdx.x;
  if (i < n) p[i] = 0u;
}

__global__ void count_rows_kernel(const int* __restrict__ row, unsigned int* __restrict__ cnt, int E) {
  int e = blockIdx.x * blockDim.x + threadIdx.x;
  if (e < E) atomicAdd(&cnt[row[e]], 1u);
}

// Single-block exclusive scan (1024 threads, Hillis-Steele per 1024-tile + carry)
__global__ __launch_bounds__(1024) void scan_kernel(const unsigned int* __restrict__ cnt,
                                                    unsigned int* __restrict__ rowptr, int n) {
  __shared__ unsigned int sh[1024];
  __shared__ unsigned int carry;
  if (threadIdx.x == 0) carry = 0u;
  __syncthreads();
  for (int base = 0; base < n; base += 1024) {
    int i = base + (int)threadIdx.x;
    unsigned int v = (i < n) ? cnt[i] : 0u;
    sh[threadIdx.x] = v;
    __syncthreads();
    for (int off = 1; off < 1024; off <<= 1) {
      unsigned int t = (threadIdx.x >= (unsigned)off) ? sh[threadIdx.x - off] : 0u;
      __syncthreads();
      sh[threadIdx.x] += t;
      __syncthreads();
    }
    if (i < n) rowptr[i] = carry + sh[threadIdx.x] - v;   // exclusive
    __syncthreads();
    if (threadIdx.x == 0) carry += sh[1023];
    __syncthreads();
  }
  if (threadIdx.x == 0) rowptr[n] = carry;
}

__global__ void copy_u32_kernel(const unsigned int* __restrict__ a, unsigned int* __restrict__ b, int n) {
  int i = blockIdx.x * blockDim.x + threadIdx.x;
  if (i < n) b[i] = a[i];
}

// Place each edge into its destination row; precompute norm = dinv[r]*dinv[s].
__global__ void scatter_edges_kernel(const int* __restrict__ row, const int* __restrict__ col,
                                     const float* __restrict__ dinv,
                                     unsigned int* __restrict__ cursor,
                                     int* __restrict__ esrc, float* __restrict__ enorm, int E) {
  int e = blockIdx.x * blockDim.x + threadIdx.x;
  if (e >= E) return;
  int r = row[e], s = col[e];
  unsigned int pos = atomicAdd(&cursor[r], 1u);
  esrc[pos]  = s;
  enorm[pos] = dinv[r] * dinv[s];
}

// ---------------------------------------------------------------------------
// GEMM1: H[M x 128] = Xbf[M x 512](bf16) * W1 (fragment-ordered bf16 panel)
// Wave owns 16 rows; A (16 K-frags) register-resident; inner = 2 vmem + 1 wmma.
// ---------------------------------------------------------------------------
__global__ __launch_bounds__(256) void gemm1_wmma_kernel(
    const unsigned short* __restrict__ Xbf, const uint4* __restrict__ Bfr,
    float* __restrict__ H, int M) {
  const int wave = threadIdx.x >> 5;
  const int lane = threadIdx.x & 31;
  const int m0 = (blockIdx.x * 8 + wave) * 16;
  if (m0 >= M) return;                 // wave-uniform => EXEC all-ones (WMMA req.)

  const int mr = lane & 15;
  const int hi = lane >> 4;
  const unsigned short* __restrict__ xrow = Xbf + (size_t)(m0 + mr) * 512;

  BFrag a[16];
#pragma unroll
  for (int kb = 0; kb < 16; ++kb) {
    a[kb].q[0] = *(const uint4*)(xrow + kb * 32 + hi * 16);
    a[kb].q[1] = *(const uint4*)(xrow + kb * 32 + hi * 16 + 8);
  }

  const uint4* __restrict__ blane = Bfr + (size_t)lane * 2;
#pragma unroll
  for (int nt = 0; nt < 8; ++nt) {
    v8f acc = {};
#pragma unroll
    for (int kb = 0; kb < 16; ++kb) {
      BFrag b;
      const uint4* bp = blane + (size_t)(nt * 16 + kb) * 64;
      b.q[0] = bp[0];
      b.q[1] = bp[1];
      acc = __builtin_amdgcn_wmma_f32_16x16x32_bf16(false, a[kb].v, false, b.v,
                                                    (short)0, acc, false, false);
    }
    float* hrow = H + (size_t)(m0 + hi * 8) * 128 + nt * 16 + mr;
#pragma unroll
    for (int r = 0; r < 8; ++r) hrow[(size_t)r * 128] = acc[r];
  }
}

// ---------------------------------------------------------------------------
// GEMM2: H2[M x 40] = Habf[M x 128](bf16, relu pre-applied) * W2 (padded N=48)
// ---------------------------------------------------------------------------
__global__ __launch_bounds__(256) void gemm2_wmma_kernel(
    const unsigned short* __restrict__ Habf, const uint4* __restrict__ Bfr,
    float* __restrict__ H2, int M, int C) {
  const int wave = threadIdx.x >> 5;
  const int lane = threadIdx.x & 31;
  const int m0 = (blockIdx.x * 8 + wave) * 16;
  if (m0 >= M) return;

  const int mr = lane & 15;
  const int hi = lane >> 4;
  const unsigned short* __restrict__ arow = Habf + (size_t)(m0 + mr) * 128;

  BFrag a[4];
#pragma unroll
  for (int kb = 0; kb < 4; ++kb) {
    a[kb].q[0] = *(const uint4*)(arow + kb * 32 + hi * 16);
    a[kb].q[1] = *(const uint4*)(arow + kb * 32 + hi * 16 + 8);
  }

  const uint4* __restrict__ blane = Bfr + (size_t)lane * 2;
#pragma unroll
  for (int nt = 0; nt < 3; ++nt) {
    v8f acc = {};
#pragma unroll
    for (int kb = 0; kb < 4; ++kb) {
      BFrag b;
      const uint4* bp = blane + (size_t)(nt * 4 + kb) * 64;
      b.q[0] = bp[0];
      b.q[1] = bp[1];
      acc = __builtin_amdgcn_wmma_f32_16x16x32_bf16(false, a[kb].v, false, b.v,
                                                    (short)0, acc, false, false);
    }
    int colg = nt * 16 + mr;
    if (colg < C) {
      float* orow = H2 + (size_t)(m0 + hi * 8) * C + colg;
#pragma unroll
      for (int r = 0; r < 8; ++r) orow[(size_t)r * C] = acc[r];
    }
  }
}

// ---------------------------------------------------------------------------
// Gather aggregation (CSR): one wave per destination node.
// out[i,f] = bias[f] + dinv[i]^2*h[i,f] + sum_p enorm[p]*h[esrc[p],f]
// ---------------------------------------------------------------------------
// F == 128: lane owns 4 contiguous floats (float4), fully coalesced 512B rows.
__global__ __launch_bounds__(256) void agg_gather128_kernel(
    const float* __restrict__ h, const unsigned int* __restrict__ rowptr,
    const int* __restrict__ esrc, const float* __restrict__ enorm,
    const float* __restrict__ dinv, const float* __restrict__ bias,
    float* __restrict__ out, int n) {
  const int wave = threadIdx.x >> 5;
  const int lane = threadIdx.x & 31;
  const int i = blockIdx.x * 8 + wave;
  if (i >= n) return;
  const int c = lane * 4;

  float d = dinv[i];
  float dd = d * d;
  float4 self = *(const float4*)(h + (size_t)i * 128 + c);
  float4 bv   = *(const float4*)(bias + c);
  float4 acc;
  acc.x = bv.x + dd * self.x;
  acc.y = bv.y + dd * self.y;
  acc.z = bv.z + dd * self.z;
  acc.w = bv.w + dd * self.w;

  unsigned int p0 = rowptr[i], p1 = rowptr[i + 1];
  for (unsigned int p = p0; p < p1; ++p) {
    int s = esrc[p];          // wave-uniform load (coalesces to one request)
    float nm = enorm[p];
    float4 hv = *(const float4*)(h + (size_t)s * 128 + c);
    acc.x += nm * hv.x;
    acc.y += nm * hv.y;
    acc.z += nm * hv.z;
    acc.w += nm * hv.w;
  }
  *(float4*)(out + (size_t)i * 128 + c) = acc;
}

// Small F (<= 64): lane owns f = lane and f = lane + 32.
__global__ __launch_bounds__(256) void agg_gather_small_kernel(
    const float* __restrict__ h, const unsigned int* __restrict__ rowptr,
    const int* __restrict__ esrc, const float* __restrict__ enorm,
    const float* __restrict__ dinv, const float* __restrict__ bias,
    float* __restrict__ out, int n, int F) {
  const int wave = threadIdx.x >> 5;
  const int lane = threadIdx.x & 31;
  const int i = blockIdx.x * 8 + wave;
  if (i >= n) return;
  const int f0 = lane, f1 = lane + 32;
  const bool h0 = f0 < F, h1 = f1 < F;

  float d = dinv[i];
  float dd = d * d;
  float a0 = h0 ? (bias[f0] + dd * h[(size_t)i * F + f0]) : 0.f;
  float a1 = h1 ? (bias[f1] + dd * h[(size_t)i * F + f1]) : 0.f;

  unsigned int p0 = rowptr[i], p1 = rowptr[i + 1];
  for (unsigned int p = p0; p < p1; ++p) {
    int s = esrc[p];
    float nm = enorm[p];
    if (h0) a0 += nm * h[(size_t)s * F + f0];
    if (h1) a1 += nm * h[(size_t)s * F + f1];
  }
  if (h0) out[(size_t)i * F + f0] = a0;
  if (h1) out[(size_t)i * F + f1] = a1;
}

// ---------------------------------------------------------------------------
// Launch
// ---------------------------------------------------------------------------
extern "C" void kernel_launch(void* const* d_in, const int* in_sizes, int n_in,
                              void* d_out, int out_size, void* d_ws, size_t ws_size,
                              hipStream_t stream) {
  const float* X  = (const float*)d_in[0];
  const int*   EI = (const int*)d_in[1];
  const float* W1 = (const float*)d_in[2];
  const float* B1 = (const float*)d_in[3];
  const float* W2 = (const float*)d_in[4];
  const float* B2 = (const float*)d_in[5];

  const int E   = in_sizes[1] / 2;        // 800000
  const int HID = in_sizes[3];            // 128
  const int C   = in_sizes[5];            // 40
  const int FIN = in_sizes[2] / HID;      // 512
  const int N   = in_sizes[0] / FIN;      // 50000
  const int CP  = ((C + 15) / 16) * 16;   // 48

  const int NT1 = HID / 16, KB1 = FIN / 32;   // 8, 16
  const int NT2 = CP / 16,  KB2 = HID / 32;   // 3, 4

  const int* erow = EI;                   // destinations
  const int* ecol = EI + E;               // sources

  // Workspace carve-up. Aliases: habf reuses xbf (dead after GEMM1),
  // h2 reuses h1 (dead after layer-1 aggregation).
  auto alignup = [](size_t x) { return (x + 255) & ~(size_t)255; };
  char* ws = (char*)d_ws;
  float* dinv          = (float*)ws;          ws += alignup((size_t)N * sizeof(float));
  float* h1            = (float*)ws;          ws += alignup((size_t)N * HID * sizeof(float));
  float* hagg          = (float*)ws;          ws += alignup((size_t)N * HID * sizeof(float));
  unsigned short* xbf  = (unsigned short*)ws; ws += alignup((size_t)N * FIN * sizeof(unsigned short));
  unsigned short* w1fr = (unsigned short*)ws; ws += alignup((size_t)NT1 * KB1 * 512 * sizeof(unsigned short));
  unsigned short* w2fr = (unsigned short*)ws; ws += alignup((size_t)NT2 * KB2 * 512 * sizeof(unsigned short));
  unsigned int* cnt    = (unsigned int*)ws;   ws += alignup((size_t)N * sizeof(unsigned int));
  unsigned int* rowptr = (unsigned int*)ws;   ws += alignup((size_t)(N + 1) * sizeof(unsigned int));
  unsigned int* cursor = (unsigned int*)ws;   ws += alignup((size_t)N * sizeof(unsigned int));
  int*   esrc          = (int*)ws;            ws += alignup((size_t)E * sizeof(int));
  float* enorm         = (float*)ws;          ws += alignup((size_t)E * sizeof(float));
  unsigned short* habf = xbf;                 // alias
  float* h2  = h1;                            // alias
  float* out = (float*)d_out;

  const int TPB = 256;

  // 1) degree (self-loop baked in as init=1) -> dinv = rsqrt(deg)
  fill1_kernel<<<(N + TPB - 1) / TPB, TPB, 0, stream>>>(dinv, N);
  deg_count_kernel<<<(E + TPB - 1) / TPB, TPB, 0, stream>>>(ecol, dinv, E);
  rsqrt_kernel<<<(N + TPB - 1) / TPB, TPB, 0, stream>>>(dinv, N);

  // 2) CSR by destination + per-edge norms
  zero_u32_kernel<<<(N + TPB - 1) / TPB, TPB, 0, stream>>>(cnt, N);
  count_rows_kernel<<<(E + TPB - 1) / TPB, TPB, 0, stream>>>(erow, cnt, E);
  scan_kernel<<<1, 1024, 0, stream>>>(cnt, rowptr, N);
  copy_u32_kernel<<<(N + TPB - 1) / TPB, TPB, 0, stream>>>(rowptr, cursor, N);
  scatter_edges_kernel<<<(E + TPB - 1) / TPB, TPB, 0, stream>>>(erow, ecol, dinv, cursor,
                                                                esrc, enorm, E);

  // 3) X -> bf16 ; weights -> fragment-ordered bf16 panels
  {
    int n4 = (N * FIN) / 4;
    cvt_bf16x4_kernel<<<(n4 + TPB - 1) / TPB, TPB, 0, stream>>>(X, xbf, n4);
  }
  build_bfrag_kernel<<<(NT1 * KB1 * 512 + TPB - 1) / TPB, TPB, 0, stream>>>(W1, w1fr, HID, NT1, KB1);
  build_bfrag_kernel<<<(NT2 * KB2 * 512 + TPB - 1) / TPB, TPB, 0, stream>>>(W2, w2fr, C, NT2, KB2);

  // 4) GEMM1: h1 = X @ W1
  {
    int mtiles = (N + 15) / 16;                    // 3125 (N % 16 == 0)
    gemm1_wmma_kernel<<<(mtiles + 7) / 8, TPB, 0, stream>>>(xbf, (const uint4*)w1fr, h1, N);
  }

  // 5) layer-1 aggregation (gather): hagg = b1 + D^-1/2 (A+I) D^-1/2 h1
  agg_gather128_kernel<<<(N + 7) / 8, TPB, 0, stream>>>(h1, rowptr, esrc, enorm, dinv, B1, hagg, N);

  // 6) relu + bf16 (habf aliases xbf)
  {
    int n4 = (N * HID) / 4;
    cvt_relu_bf16x4_kernel<<<(n4 + TPB - 1) / TPB, TPB, 0, stream>>>(hagg, habf, n4);
  }

  // 7) GEMM2: h2 = relu(hagg) @ W2
  {
    int mtiles = (N + 15) / 16;
    gemm2_wmma_kernel<<<(mtiles + 7) / 8, TPB, 0, stream>>>(habf, (const uint4*)w2fr, h2, N, C);
  }

  // 8) layer-2 aggregation (gather) into d_out
  agg_gather_small_kernel<<<(N + 7) / 8, TPB, 0, stream>>>(h2, rowptr, esrc, enorm, dinv, B2, out, N, C);
}
